// TextCrossAttention_72507637891281
// MI455X (gfx1250) — matrix operand order repaired and checked by reading, not compiled, not run on hardware
//
#include <hip/hip_runtime.h>
#include <hip/hip_bf16.h>
#include <cstdint>
#include <cstddef>

// ---------------------------------------------------------------- problem dims
#define BB  8
#define CC  768
#define HH  64
#define WW  64
#define HWP (HH*WW)          // 4096 pixels
#define LL  256
#define NHH 8
#define HDD 96               // head dim
#define EPSF 1e-5f

typedef __attribute__((ext_vector_type(16))) _Float16 v16h;
typedef __attribute__((ext_vector_type(8)))  _Float16 v8h;
typedef __attribute__((ext_vector_type(8)))  float    v8f;
typedef int v4i __attribute__((ext_vector_type(4)));

union V16U { v16h v; v8h h[2]; };

// CDNA5 async global->LDS path (gfx1250): GLOBAL_LOAD_ASYNC_TO_LDS_B128,
// tracked with ASYNCcnt. Guarded so we fall back cleanly if the toolchain
// lacks the builtins.
#if defined(__HIP_DEVICE_COMPILE__) && \
    __has_builtin(__builtin_amdgcn_global_load_async_to_lds_b128) && \
    __has_builtin(__builtin_amdgcn_s_wait_asynccnt)
#define HAVE_ASYNC_LDS 1
#else
#define HAVE_ASYNC_LDS 0
#endif

#if HAVE_ASYNC_LDS
__device__ __forceinline__ void async_cp16(const _Float16* g, _Float16* l) {
  __builtin_amdgcn_global_load_async_to_lds_b128(
      (__attribute__((address_space(1))) v4i*)g,
      (__attribute__((address_space(3))) v4i*)l, 0, 0);
}
#endif

// ---------------------------------------------------------------- WMMA GEMM core
// D[M,N] = A[M,K] * W[N,K]^T   (both row-major, K contiguous)
// Block: 256 threads = 8 waves arranged 4x2; wave tile 32x32 (2x2 WMMA tiles);
// block tile 128x64, K step 32. LDS row stride padded to 40 halves.
//
// NOTE on bounds: every launch uses M % 128 == 0 and K % 32 == 0, and W-operand
// over-reads (only PV, N=96 in a 64-wide tile) land in adjacent valid workspace
// and feed accumulator columns that the epilogue never stores. So the mainloop
// performs NO bounds checks; only the epilogue guards stores.
#define BM 128
#define BN 64
#define BK 32
#define LDS_STRIDE 40
#define A_TILE (BM * LDS_STRIDE)
#define B_TILE (BN * LDS_STRIDE)

__device__ __forceinline__ void gemm_compute_step(
    const _Float16* ldsA, const _Float16* ldsB,
    int wm, int wn, int hsel, int l16, v8f acc[2][2])
{
  V16U bf[2];
#pragma unroll
  for (int tn = 0; tn < 2; ++tn) {
    int n = wn * 32 + tn * 16 + l16;
    const _Float16* p = &ldsB[n * LDS_STRIDE + hsel * 16];
    bf[tn].h[0] = *(const v8h*)p;
    bf[tn].h[1] = *(const v8h*)(p + 8);
  }
#pragma unroll
  for (int tm = 0; tm < 2; ++tm) {
    int r = wm * 32 + tm * 16 + l16;
    const _Float16* pa = &ldsA[r * LDS_STRIDE + hsel * 16];
    V16U af;
    af.h[0] = *(const v8h*)pa;
    af.h[1] = *(const v8h*)(pa + 8);
#pragma unroll
    for (int tn = 0; tn < 2; ++tn) {
      acc[tm][tn] = __builtin_amdgcn_wmma_f32_16x16x32_f16(
          false, af.v, false, bf[tn].v, (short)0, acc[tm][tn], false, false);
    }
  }
}

#if HAVE_ASYNC_LDS
__device__ __forceinline__ void stage_tile_async(
    const _Float16* __restrict__ A, int lda,
    const _Float16* __restrict__ Bw, int ldb,
    int rowBlk, int colBlk, int k0,
    _Float16* ldsA, _Float16* ldsB, int tid)
{
#pragma unroll
  for (int it = 0; it < 2; ++it) {
    int chunk = tid + it * 256;
    int r  = chunk >> 2;
    int cc = (chunk & 3) << 3;
    async_cp16(A + (size_t)(rowBlk + r) * lda + k0 + cc, &ldsA[r * LDS_STRIDE + cc]);
  }
  {
    int r  = tid >> 2;
    int cc = (tid & 3) << 3;
    async_cp16(Bw + (size_t)(colBlk + r) * ldb + k0 + cc, &ldsB[r * LDS_STRIDE + cc]);
  }
}
#endif

__device__ __forceinline__ void gemm_tiles(
    const _Float16* __restrict__ A, int lda,
    const _Float16* __restrict__ Bw, int ldb,
    int K, int rowBlk, int colBlk,
    _Float16* ldsA, _Float16* ldsB, v8f acc[2][2])
{
  const int tid  = threadIdx.x;
  const int lane = tid & 31;
  const int wid  = tid >> 5;
  const int wm   = wid >> 1;       // 0..3
  const int wn   = wid & 1;        // 0..1
  const int hsel = lane >> 4;      // 0/1 (lane half)
  const int l16  = lane & 15;

#if HAVE_ASYNC_LDS
  // Double-buffered async pipeline: fetch tile k+1 while WMMA runs on tile k.
  stage_tile_async(A, lda, Bw, ldb, rowBlk, colBlk, 0, ldsA, ldsB, tid);
  int cur = 0;
  for (int k0 = 0; k0 < K; k0 += BK) {
    __builtin_amdgcn_s_wait_asynccnt(0);
    __syncthreads();
    if (k0 + BK < K) {
      stage_tile_async(A, lda, Bw, ldb, rowBlk, colBlk, k0 + BK,
                       ldsA + (cur ^ 1) * A_TILE, ldsB + (cur ^ 1) * B_TILE, tid);
    }
    gemm_compute_step(ldsA + cur * A_TILE, ldsB + cur * B_TILE,
                      wm, wn, hsel, l16, acc);
    cur ^= 1;
  }
#else
  for (int k0 = 0; k0 < K; k0 += BK) {
#pragma unroll
    for (int it = 0; it < 2; ++it) {
      int chunk = tid + it * 256;
      int r  = chunk >> 2;
      int cc = (chunk & 3) << 3;
      v8h val = *(const v8h*)(A + (size_t)(rowBlk + r) * lda + k0 + cc);
      *(v8h*)(&ldsA[r * LDS_STRIDE + cc]) = val;
    }
    {
      int r  = tid >> 2;
      int cc = (tid & 3) << 3;
      v8h val = *(const v8h*)(Bw + (size_t)(colBlk + r) * ldb + k0 + cc);
      *(v8h*)(&ldsB[r * LDS_STRIDE + cc]) = val;
    }
    __syncthreads();
    gemm_compute_step(ldsA, ldsB, wm, wn, hsel, l16, acc);
    __syncthreads();
  }
#endif
}

#if HAVE_ASYNC_LDS
#define DECL_GEMM_LDS \
  __shared__ _Float16 ldsA[2 * A_TILE]; \
  __shared__ _Float16 ldsB[2 * B_TILE];
#else
#define DECL_GEMM_LDS \
  __shared__ _Float16 ldsA[A_TILE]; \
  __shared__ _Float16 ldsB[B_TILE];
#endif

// Generic batched GEMM: optional f32 out (with alpha), optional f16 out, optional bias.
__global__ __launch_bounds__(256) void wmma_gemm_kernel(
    const _Float16* __restrict__ A, int lda, long long aSo, long long aSi,
    const _Float16* __restrict__ Bw, int ldb, long long bSo, long long bSi,
    const float* __restrict__ bias,
    float* __restrict__ out32, int ldo32, long long oSo32, long long oSi32,
    _Float16* __restrict__ out16, int ldo16, long long oSo16, long long oSi16,
    int M, int N, int K, int batchInner, float alpha)
{
  DECL_GEMM_LDS
  const int z  = blockIdx.z;
  const int zo = z / batchInner;
  const int zi = z - zo * batchInner;
  const _Float16* Ab = A  + zo * aSo + zi * aSi;
  const _Float16* Wb = Bw + zo * bSo + zi * bSi;
  const int rowBlk = blockIdx.y * BM;
  const int colBlk = blockIdx.x * BN;

  v8f acc[2][2];
#pragma unroll
  for (int a = 0; a < 2; ++a)
#pragma unroll
    for (int b = 0; b < 2; ++b)
#pragma unroll
      for (int i = 0; i < 8; ++i) acc[a][b][i] = 0.f;

  gemm_tiles(Ab, lda, Wb, ldb, K, rowBlk, colBlk, ldsA, ldsB, acc);

  const int lane = threadIdx.x & 31;
  const int wid  = threadIdx.x >> 5;
  const int wm = wid >> 1, wn = wid & 1;
#pragma unroll
  for (int tm = 0; tm < 2; ++tm)
#pragma unroll
    for (int tn = 0; tn < 2; ++tn) {
      int col  = colBlk + wn * 32 + tn * 16 + (lane & 15);
      int row0 = rowBlk + wm * 32 + tm * 16 + (lane >> 4) * 8;
      if (col >= N) continue;
      float bv = bias ? bias[col] : 0.f;
#pragma unroll
      for (int i = 0; i < 8; ++i) {
        int row = row0 + i;
        if (row >= M) break;
        float v = acc[tm][tn][i] * alpha + bv;
        if (out32) out32[oSo32 * zo + oSi32 * zi + (size_t)row * ldo32 + col] = v;
        if (out16) out16[oSo16 * zo + oSi16 * zi + (size_t)row * ldo16 + col] = (_Float16)v;
      }
    }
}

// Final proj GEMM with fused epilogue: fused[b,c,p] = vt32[r,c] + (acc+bias[c])*gate[b,c]
__global__ __launch_bounds__(256) void wmma_gemm_proj_fused_kernel(
    const _Float16* __restrict__ A, int lda,
    const _Float16* __restrict__ Bw, int ldb,
    const float* __restrict__ bias,
    const float* __restrict__ gate,     // (B, C)
    const float* __restrict__ vt32,     // (B*HW, C)
    float* __restrict__ fused,          // (B, C, HW)
    int M, int N, int K)
{
  DECL_GEMM_LDS
  const int rowBlk = blockIdx.y * BM;
  const int colBlk = blockIdx.x * BN;

  v8f acc[2][2];
#pragma unroll
  for (int a = 0; a < 2; ++a)
#pragma unroll
    for (int b = 0; b < 2; ++b)
#pragma unroll
      for (int i = 0; i < 8; ++i) acc[a][b][i] = 0.f;

  gemm_tiles(A, lda, Bw, ldb, K, rowBlk, colBlk, ldsA, ldsB, acc);

  const int lane = threadIdx.x & 31;
  const int wid  = threadIdx.x >> 5;
  const int wm = wid >> 1, wn = wid & 1;
#pragma unroll
  for (int tm = 0; tm < 2; ++tm)
#pragma unroll
    for (int tn = 0; tn < 2; ++tn) {
      int col  = colBlk + wn * 32 + tn * 16 + (lane & 15);
      int row0 = rowBlk + wm * 32 + tm * 16 + (lane >> 4) * 8;
      if (col >= N) continue;
      float bv = bias[col];
      // rows within a 128-row block never cross a batch boundary (HW % 128 == 0)
      int bIdx = row0 / HWP;
      float g  = gate[bIdx * CC + col];
#pragma unroll
      for (int i = 0; i < 8; ++i) {
        int row = row0 + i;
        if (row >= M) break;
        int p = row - bIdx * HWP;
        float v = (acc[tm][tn][i] + bv) * g + vt32[(size_t)row * CC + col];
        fused[(size_t)bIdx * CC * HWP + (size_t)col * HWP + p] = v;
      }
    }
}

// ---------------------------------------------------------------- pre/post kernels

// LayerNorm over C of visual feat (B,C,H,W) -> vt32 (B*HW, C) and vt16
__global__ __launch_bounds__(256) void ln_visual_kernel(
    const float* __restrict__ x, const float* __restrict__ g,
    const float* __restrict__ bta, float* __restrict__ vt32,
    _Float16* __restrict__ vt16)
{
  __shared__ float sred[256];
  const int pix = blockIdx.x;            // 0..B*HW-1
  const int b = pix / HWP, p = pix - b * HWP;
  const float* base = x + (size_t)b * CC * HWP + p;
  const int tid = threadIdx.x;

  float vals[3];
  float s = 0.f, s2 = 0.f;
#pragma unroll
  for (int j = 0; j < 3; ++j) {
    int c = tid + 256 * j;
    float v = base[(size_t)c * HWP];
    vals[j] = v; s += v; s2 += v * v;
  }
  sred[tid] = s; __syncthreads();
  for (int off = 128; off > 0; off >>= 1) { if (tid < off) sred[tid] += sred[tid + off]; __syncthreads(); }
  float S = sred[0]; __syncthreads();
  sred[tid] = s2; __syncthreads();
  for (int off = 128; off > 0; off >>= 1) { if (tid < off) sred[tid] += sred[tid + off]; __syncthreads(); }
  float S2 = sred[0];

  float mean = S * (1.f / CC);
  float var  = S2 * (1.f / CC) - mean * mean;
  float rs   = rsqrtf(var + EPSF);
#pragma unroll
  for (int j = 0; j < 3; ++j) {
    int c = tid + 256 * j;
    float o = (vals[j] - mean) * rs * g[c] + bta[c];
    vt32[(size_t)pix * CC + c] = o;
    vt16[(size_t)pix * CC + c] = (_Float16)o;
  }
}

// LayerNorm over C of text tokens (B,L,C) -> tn16 (B*L, C)
__global__ __launch_bounds__(256) void ln_text_kernel(
    const float* __restrict__ x, const float* __restrict__ g,
    const float* __restrict__ bta, _Float16* __restrict__ out16)
{
  __shared__ float sred[256];
  const int row = blockIdx.x;            // 0..B*L-1
  const float* base = x + (size_t)row * CC;
  const int tid = threadIdx.x;

  float vals[3];
  float s = 0.f, s2 = 0.f;
#pragma unroll
  for (int j = 0; j < 3; ++j) {
    int c = tid + 256 * j;
    float v = base[c];
    vals[j] = v; s += v; s2 += v * v;
  }
  sred[tid] = s; __syncthreads();
  for (int off = 128; off > 0; off >>= 1) { if (tid < off) sred[tid] += sred[tid + off]; __syncthreads(); }
  float S = sred[0]; __syncthreads();
  sred[tid] = s2; __syncthreads();
  for (int off = 128; off > 0; off >>= 1) { if (tid < off) sred[tid] += sred[tid + off]; __syncthreads(); }
  float S2 = sred[0];

  float mean = S * (1.f / CC);
  float var  = S2 * (1.f / CC) - mean * mean;
  float rs   = rsqrtf(var + EPSF);
#pragma unroll
  for (int j = 0; j < 3; ++j) {
    int c = tid + 256 * j;
    out16[(size_t)row * CC + c] = (_Float16)((vals[j] - mean) * rs * g[c] + bta[c]);
  }
}

__global__ __launch_bounds__(256) void f32_to_f16_kernel(
    const float* __restrict__ in, _Float16* __restrict__ out, int n)
{
  int i = blockIdx.x * 256 + threadIdx.x;
  if (i < n) out[i] = (_Float16)in[i];
}

// in-place row softmax over L=256, also emit f16 copy; one wave per row
__global__ __launch_bounds__(256) void softmax_kernel(
    float* __restrict__ logits, _Float16* __restrict__ w16)
{
  const int wid = threadIdx.x >> 5, lane = threadIdx.x & 31;
  const long long row = (long long)blockIdx.x * 8 + wid;
  float* p = logits + row * LL;

  float v[8];
  float m = -1e30f;
#pragma unroll
  for (int j = 0; j < 8; ++j) { v[j] = p[lane + 32 * j]; m = fmaxf(m, v[j]); }
#pragma unroll
  for (int off = 16; off > 0; off >>= 1) m = fmaxf(m, __shfl_xor(m, off, 32));
  float s = 0.f;
#pragma unroll
  for (int j = 0; j < 8; ++j) { v[j] = __expf(v[j] - m); s += v[j]; }
#pragma unroll
  for (int off = 16; off > 0; off >>= 1) s += __shfl_xor(s, off, 32);
  float inv = 1.f / s;
#pragma unroll
  for (int j = 0; j < 8; ++j) {
    float o = v[j] * inv;
    p[lane + 32 * j] = o;
    w16[row * LL + lane + 32 * j] = (_Float16)o;
  }
}

// vT[(b*NH+h)*HD*L + n*L + k] = kv16[b*L*1536 + k*1536 + 768 + h*HD + n]
__global__ __launch_bounds__(256) void v_transpose_kernel(
    const _Float16* __restrict__ kv16, _Float16* __restrict__ vT)
{
  int idx = blockIdx.x * 256 + threadIdx.x;
  const int total = BB * NHH * HDD * LL;
  if (idx >= total) return;
  int k = idx & (LL - 1);
  int n = (idx / LL) % HDD;
  int z = idx / (LL * HDD);
  int b = z / NHH, h = z - b * NHH;
  vT[idx] = kv16[(size_t)b * LL * (2 * CC) + (size_t)k * (2 * CC) + CC + h * HDD + n];
}

// ttmean[b,c] = mean over L of tt16[b,l,c]
__global__ __launch_bounds__(256) void ttmean_kernel(
    const _Float16* __restrict__ tt16, float* __restrict__ ttmean)
{
  int idx = blockIdx.x * 256 + threadIdx.x;
  if (idx >= BB * CC) return;
  int b = idx / CC, c = idx - b * CC;
  float s = 0.f;
  for (int l = 0; l < LL; ++l) s += (float)tt16[(size_t)b * LL * CC + (size_t)l * CC + c];
  ttmean[idx] = s * (1.f / LL);
}

// gate[b,n] = sigmoid( dot(ttmean[b,:], gate_w[n,:]) + gate_b[n] )
__global__ __launch_bounds__(256) void gate_kernel(
    const float* __restrict__ ttmean, const float* __restrict__ gw,
    const float* __restrict__ gb, float* __restrict__ gate)
{
  int idx = blockIdx.x * 256 + threadIdx.x;
  if (idx >= BB * CC) return;
  int b = idx / CC, n = idx - b * CC;
  const float* m = ttmean + b * CC;
  const float* w = gw + (size_t)n * CC;
  float s = 0.f;
  for (int c = 0; c < CC; ++c) s += m[c] * w[c];
  s += gb[n];
  gate[idx] = 1.f / (1.f + __expf(-s));
}

// ---------------------------------------------------------------- host launcher
static inline size_t alignup256(size_t x) { return (x + 255) & ~(size_t)255; }

extern "C" void kernel_launch(void* const* d_in, const int* in_sizes, int n_in,
                              void* d_out, int out_size, void* d_ws, size_t ws_size,
                              hipStream_t stream) {
  (void)in_sizes; (void)n_in; (void)out_size; (void)ws_size;
  const float* visual   = (const float*)d_in[0];
  const float* text     = (const float*)d_in[1];
  const float* vn_g     = (const float*)d_in[2];
  const float* vn_b     = (const float*)d_in[3];
  const float* tn_g     = (const float*)d_in[4];
  const float* tn_b     = (const float*)d_in[5];
  const float* t2v_w    = (const float*)d_in[6];
  const float* t2v_b    = (const float*)d_in[7];
  const float* inproj_w = (const float*)d_in[8];
  const float* inproj_b = (const float*)d_in[9];
  const float* out_w    = (const float*)d_in[10];
  const float* out_b    = (const float*)d_in[11];
  const float* gate_w   = (const float*)d_in[12];
  const float* gate_b   = (const float*)d_in[13];
  const float* proj_w   = (const float*)d_in[14];
  const float* proj_b   = (const float*)d_in[15];

  float* fused = (float*)d_out;                               // (B, C, HW)
  float* attnw = (float*)d_out + (size_t)BB * CC * HWP;       // (B, NH, HW, L)

  // --- workspace carve ---
  char* w = (char*)d_ws;
  auto carve = [&](size_t bytes) { char* p = w; w += alignup256(bytes); return p; };
  _Float16* W16_T2V    = (_Float16*)carve((size_t)CC * CC * 2);
  _Float16* W16_INPROJ = (_Float16*)carve((size_t)3 * CC * CC * 2);
  _Float16* W16_OUT    = (_Float16*)carve((size_t)CC * CC * 2);
  _Float16* W16_PROJ   = (_Float16*)carve((size_t)CC * CC * 2);
  float*    VT32       = (float*)   carve((size_t)BB * HWP * CC * 4);
  _Float16* VT16       = (_Float16*)carve((size_t)BB * HWP * CC * 2);
  _Float16* TN16       = (_Float16*)carve((size_t)BB * LL * CC * 2);
  _Float16* TT16       = (_Float16*)carve((size_t)BB * LL * CC * 2);
  _Float16* Q16        = (_Float16*)carve((size_t)BB * HWP * CC * 2);
  _Float16* KV16       = (_Float16*)carve((size_t)BB * LL * 2 * CC * 2);
  _Float16* VTT16      = (_Float16*)carve((size_t)BB * NHH * HDD * LL * 2);
  _Float16* ATTNW16    = (_Float16*)carve((size_t)BB * NHH * HWP * LL * 2);
  _Float16* CTX16      = (_Float16*)carve((size_t)BB * HWP * CC * 2);
  _Float16* Y116       = (_Float16*)carve((size_t)BB * HWP * CC * 2);
  float*    TTMEAN     = (float*)   carve((size_t)BB * CC * 4);
  float*    GATE       = (float*)   carve((size_t)BB * CC * 4);

  const float qk_scale = 0.10206207261596577f;  // 1/sqrt(96)

  // 1) weight conversions f32 -> f16
  {
    int n1 = CC * CC, n3 = 3 * CC * CC;
    f32_to_f16_kernel<<<(n1 + 255) / 256, 256, 0, stream>>>(t2v_w,    W16_T2V,    n1);
    f32_to_f16_kernel<<<(n3 + 255) / 256, 256, 0, stream>>>(inproj_w, W16_INPROJ, n3);
    f32_to_f16_kernel<<<(n1 + 255) / 256, 256, 0, stream>>>(out_w,    W16_OUT,    n1);
    f32_to_f16_kernel<<<(n1 + 255) / 256, 256, 0, stream>>>(proj_w,   W16_PROJ,   n1);
  }

  // 2) layernorms
  ln_visual_kernel<<<BB * HWP, 256, 0, stream>>>(visual, vn_g, vn_b, VT32, VT16);
  ln_text_kernel<<<BB * LL, 256, 0, stream>>>(text, tn_g, tn_b, TN16);

  // 3) t2v: TT16 = TN16 @ t2v_w^T + t2v_b     (M=2048, N=768, K=768)
  {
    dim3 g(CC / BN, (BB * LL) / BM, 1);
    wmma_gemm_kernel<<<g, 256, 0, stream>>>(
        TN16, CC, 0, 0, W16_T2V, CC, 0, 0, t2v_b,
        nullptr, 0, 0, 0, TT16, CC, 0, 0,
        BB * LL, CC, CC, 1, 1.f);
  }
  // 4) Q: Q16 = VT16 @ wq^T + bq             (M=32768, N=768, K=768)
  {
    dim3 g(CC / BN, (BB * HWP) / BM, 1);
    wmma_gemm_kernel<<<g, 256, 0, stream>>>(
        VT16, CC, 0, 0, W16_INPROJ, CC, 0, 0, inproj_b,
        nullptr, 0, 0, 0, Q16, CC, 0, 0,
        BB * HWP, CC, CC, 1, 1.f);
  }
  // 5) KV: KV16 = TT16 @ [wk;wv]^T + [bk;bv] (M=2048, N=1536, K=768)
  {
    dim3 g((2 * CC) / BN, (BB * LL) / BM, 1);
    wmma_gemm_kernel<<<g, 256, 0, stream>>>(
        TT16, CC, 0, 0, W16_INPROJ + (size_t)CC * CC, CC, 0, 0, inproj_b + CC,
        nullptr, 0, 0, 0, KV16, 2 * CC, 0, 0,
        BB * LL, 2 * CC, CC, 1, 1.f);
  }
  // 6) QK logits -> d_out attn region (scaled); batched over (b,h)
  {
    dim3 g(LL / BN, HWP / BM, BB * NHH);
    wmma_gemm_kernel<<<g, 256, 0, stream>>>(
        Q16, CC, (long long)HWP * CC, HDD,
        KV16, 2 * CC, (long long)LL * 2 * CC, HDD,   // k part = first 768 cols of each row
        nullptr,
        attnw, LL, (long long)NHH * HWP * LL, (long long)HWP * LL,
        nullptr, 0, 0, 0,
        HWP, LL, HDD, NHH, qk_scale);
  }
  // 7) softmax in place + f16 copy
  softmax_kernel<<<(BB * NHH * HWP) / 8, 256, 0, stream>>>(attnw, ATTNW16);

  // 8) V transpose for PV
  {
    int total = BB * NHH * HDD * LL;
    v_transpose_kernel<<<(total + 255) / 256, 256, 0, stream>>>(KV16, VTT16);
  }
  // 9) PV: CTX16[b, q, h*96+d] = attnw @ v   (M=4096, N=96, K=256, batched 64)
  {
    dim3 g((HDD + BN - 1) / BN, HWP / BM, BB * NHH);
    wmma_gemm_kernel<<<g, 256, 0, stream>>>(
        ATTNW16, LL, (long long)NHH * HWP * LL, (long long)HWP * LL,
        VTT16, LL, (long long)NHH * HDD * LL, (long long)HDD * LL,
        nullptr,
        nullptr, 0, 0, 0,
        CTX16, CC, (long long)HWP * CC, HDD,
        HWP, HDD, LL, NHH, 1.f);
  }
  // 10) mha_out: Y116 = CTX16 @ out_w^T + out_b  (M=32768, N=768, K=768)
  {
    dim3 g(CC / BN, (BB * HWP) / BM, 1);
    wmma_gemm_kernel<<<g, 256, 0, stream>>>(
        CTX16, CC, 0, 0, W16_OUT, CC, 0, 0, out_b,
        nullptr, 0, 0, 0, Y116, CC, 0, 0,
        BB * HWP, CC, CC, 1, 1.f);
  }
  // 11) gate
  ttmean_kernel<<<(BB * CC + 255) / 256, 256, 0, stream>>>(TT16, TTMEAN);
  gate_kernel<<<(BB * CC + 255) / 256, 256, 0, stream>>>(TTMEAN, gate_w, gate_b, GATE);

  // 12) proj + gate + residual + NCHW transpose, fused epilogue
  {
    dim3 g(CC / BN, (BB * HWP) / BM, 1);
    wmma_gemm_proj_fused_kernel<<<g, 256, 0, stream>>>(
        Y116, CC, W16_PROJ, CC, proj_b, GATE, VT32, fused,
        BB * HWP, CC, CC);
  }
}
